// VectorQuantizerEMA_17171279249686
// MI455X (gfx1250) — matrix operand order repaired and checked by reading, not compiled, not run on hardware
//
#include <hip/hip_runtime.h>

#define NUM_CODES 1024
#define DIM 256
#define N_ROWS 65536            // 64 * 1024
#define Q_ELEMS (N_ROWS * DIM)  // 16777216
#define LDS_STRIDE 264          // bf16 elems per code row in LDS (pad for banks)
#define MAIN_BLOCKS 512         // 65536 rows / 128 rows-per-block

typedef __attribute__((ext_vector_type(16))) __bf16 v16bf;
typedef __attribute__((ext_vector_type(8)))  __bf16 v8bf;
typedef __attribute__((ext_vector_type(8)))  float  v8f;

static __device__ __forceinline__ v8f wmma_bf16(v16bf a, v16bf b, v8f c) {
  // D = A(16x32 bf16) * B(32x16 bf16) + C(16x16 f32)
  return __builtin_amdgcn_wmma_f32_16x16x32_bf16(
      /*neg_a=*/false, a, /*neg_b=*/false, b,
      /*c_mod=*/(short)0, c, /*reuse_a=*/false, /*reuse_b=*/false);
}

// ---------------------------------------------------------------------------
// Kernel 1: codebook -> bf16 hi/lo decomposition + per-code squared norms
// grid = 1024 blocks (one per code), block = 256 threads (one per dim)
// ---------------------------------------------------------------------------
__global__ void vq_prep(const float* __restrict__ emb,
                        __bf16* __restrict__ ehi, __bf16* __restrict__ elo,
                        float* __restrict__ enorm) {
  const int k = blockIdx.x;
  const int t = threadIdx.x;
  const float x = emb[k * DIM + t];
  const __bf16 h = (__bf16)x;
  const float hf = (float)h;
  const __bf16 l = (__bf16)(x - hf);
  ehi[k * DIM + t] = h;
  elo[k * DIM + t] = l;

  float sq = x * x;
  #pragma unroll
  for (int m = 1; m < 32; m <<= 1) sq += __shfl_xor(sq, m, 32);

  __shared__ float red[8];
  const int lane = t & 31, w = t >> 5;
  if (lane == 0) red[w] = sq;
  __syncthreads();
  if (t == 0) {
    float s = 0.f;
    #pragma unroll
    for (int i = 0; i < 8; ++i) s += red[i];
    enorm[k] = s;
  }
}

// ---------------------------------------------------------------------------
// Kernel 2: distances via WMMA (bf16x3), argmin, gather, loss partials
// grid = 512 blocks x 256 threads (8 waves); each wave owns 16 rows
// ---------------------------------------------------------------------------
__global__ __launch_bounds__(256) void vq_main(
    const float* __restrict__ inputs, const float* __restrict__ emb,
    const __bf16* __restrict__ ehi, const __bf16* __restrict__ elo,
    const float* __restrict__ enorm,
    float* __restrict__ out_q, float* __restrict__ out_idx,
    float* __restrict__ partials) {

  __shared__ __bf16 ldsB[2][16][LDS_STRIDE];  // [hi/lo][code][dim]
  __shared__ float wsum[8];

  const int tid  = threadIdx.x;
  const int lane = tid & 31;
  const int wave = tid >> 5;
  const int half = lane >> 4;   // 0: lanes 0-15, 1: lanes 16-31
  const int l15  = lane & 15;
  const int rowBase = blockIdx.x * 128 + wave * 16;

  // ---- Load A tile (16 rows x 256 dims) into registers in WMMA A layout ----
  // A 16x32 bf16 layout: lane L(+16) holds row L; v16bf elems 0-7 -> K=base..base+7,
  // elems 8-15 -> K=base+16..base+23, base = 32c + 8*half.
  v16bf a_hi[8], a_lo[8];
  {
    const float* rowPtr = inputs + (size_t)(rowBase + l15) * DIM;
    #pragma unroll
    for (int c = 0; c < 8; ++c) {
      const int base = 32 * c + half * 8;
      const float4 f0 = *(const float4*)(rowPtr + base);
      const float4 f1 = *(const float4*)(rowPtr + base + 4);
      const float4 f2 = *(const float4*)(rowPtr + base + 16);
      const float4 f3 = *(const float4*)(rowPtr + base + 20);
      float xs[16] = {f0.x, f0.y, f0.z, f0.w, f1.x, f1.y, f1.z, f1.w,
                      f2.x, f2.y, f2.z, f2.w, f3.x, f3.y, f3.z, f3.w};
      #pragma unroll
      for (int e = 0; e < 16; ++e) {
        const __bf16 h = (__bf16)xs[e];
        a_hi[c][e] = h;
        a_lo[c][e] = (__bf16)(xs[e] - (float)h);
      }
    }
  }

  float best[8];
  int   bidx[8];
  #pragma unroll
  for (int r = 0; r < 8; ++r) { best[r] = 3.4e38f; bidx[r] = 0; }

  // ---- Sweep 64 codebook tiles of 16 codes each ----
  for (int kt = 0; kt < 64; ++kt) {
    __syncthreads();
    {  // cooperative stage of B tile (16 codes x 256 dims, hi+lo) into LDS
      const int code = tid >> 4;          // 0..15
      const int seg  = (tid & 15) * 16;   // dim segment start (16 bf16 = 32B)
      const size_t g = (size_t)(kt * 16 + code) * DIM + seg;
      const uint4* sh = (const uint4*)(ehi + g);
      const uint4* sl = (const uint4*)(elo + g);
      uint4* dh = (uint4*)&ldsB[0][code][seg];
      uint4* dl = (uint4*)&ldsB[1][code][seg];
      dh[0] = sh[0]; dh[1] = sh[1];
      dl[0] = sl[0]; dl[1] = sl[1];
    }
    __syncthreads();

    const float en = enorm[kt * 16 + l15];

    // B 32x16 bf16 layout: column = lane&15; elems 0-7 -> K=base..base+7,
    // elems 8-15 -> K=base+16..base+23, base = 32c + 8*half (matches A).
    v8f acc0 = {}, acc1 = {}, acc2 = {};
    #pragma unroll
    for (int c = 0; c < 8; ++c) {
      const int base = 32 * c + half * 8;
      const v8bf bh0 = *(const v8bf*)&ldsB[0][l15][base];
      const v8bf bh1 = *(const v8bf*)&ldsB[0][l15][base + 16];
      const v8bf bl0 = *(const v8bf*)&ldsB[1][l15][base];
      const v8bf bl1 = *(const v8bf*)&ldsB[1][l15][base + 16];
      const v16bf bh = __builtin_shufflevector(bh0, bh1, 0,1,2,3,4,5,6,7,8,9,10,11,12,13,14,15);
      const v16bf bl = __builtin_shufflevector(bl0, bl1, 0,1,2,3,4,5,6,7,8,9,10,11,12,13,14,15);
      acc0 = wmma_bf16(a_hi[c], bh, acc0);   // hi*hi
      acc1 = wmma_bf16(a_hi[c], bl, acc1);   // hi*lo
      acc2 = wmma_bf16(a_lo[c], bh, acc2);   // lo*hi
    }

    // C layout: elem r -> row (r + 8*half), col l15
    const int kcol = kt * 16 + l15;
    #pragma unroll
    for (int r = 0; r < 8; ++r) {
      const float dot = acc0[r] + acc1[r] + acc2[r];
      const float s = en - 2.0f * dot;        // ||e||^2 - 2 x.e
      if (s < best[r]) { best[r] = s; bidx[r] = kcol; }
    }
  }

  // ---- argmin across the 16 columns held by each 16-lane group ----
  #pragma unroll
  for (int off = 1; off < 16; off <<= 1) {
    #pragma unroll
    for (int r = 0; r < 8; ++r) {
      const float ov = __shfl_xor(best[r], off, 32);
      const int   oi = __shfl_xor(bidx[r], off, 32);
      if (ov < best[r] || (ov == best[r] && oi < bidx[r])) {
        best[r] = ov; bidx[r] = oi;
      }
    }
  }
  if (l15 == 0) {
    #pragma unroll
    for (int r = 0; r < 8; ++r)
      out_idx[rowBase + half * 8 + r] = (float)bidx[r];  // exact for idx<1024
  }

  // ---- gather quantized rows (fp32 codebook) + commitment-loss partial ----
  float lsum = 0.f;
  #pragma unroll
  for (int m = 0; m < 16; ++m) {
    const int idx = __shfl(bidx[m & 7], (m < 8) ? 0 : 16, 32);
    const float4* erow = (const float4*)(emb + (size_t)idx * DIM);
    const float4* xrow = (const float4*)(inputs + (size_t)(rowBase + m) * DIM);
    float4* qrow = (float4*)(out_q + (size_t)(rowBase + m) * DIM);
    const float4 e0 = erow[lane * 2], e1 = erow[lane * 2 + 1];
    const float4 x0 = xrow[lane * 2], x1 = xrow[lane * 2 + 1];
    qrow[lane * 2]     = e0;
    qrow[lane * 2 + 1] = e1;
    float d;
    d = e0.x - x0.x; lsum += d * d;  d = e0.y - x0.y; lsum += d * d;
    d = e0.z - x0.z; lsum += d * d;  d = e0.w - x0.w; lsum += d * d;
    d = e1.x - x1.x; lsum += d * d;  d = e1.y - x1.y; lsum += d * d;
    d = e1.z - x1.z; lsum += d * d;  d = e1.w - x1.w; lsum += d * d;
  }
  #pragma unroll
  for (int m = 1; m < 32; m <<= 1) lsum += __shfl_xor(lsum, m, 32);
  if (lane == 0) wsum[wave] = lsum;
  __syncthreads();
  if (tid == 0) {
    float s = 0.f;
    #pragma unroll
    for (int i = 0; i < 8; ++i) s += wsum[i];
    partials[blockIdx.x] = s;
  }
}

// ---------------------------------------------------------------------------
// Kernel 3: deterministic fixed-order loss reduction
// ---------------------------------------------------------------------------
__global__ void vq_finish(const float* __restrict__ partials,
                          float* __restrict__ out_loss) {
  if (threadIdx.x == 0 && blockIdx.x == 0) {
    double s = 0.0;
    for (int i = 0; i < MAIN_BLOCKS; ++i) s += (double)partials[i];
    out_loss[0] = (float)(0.25 * s / (double)Q_ELEMS);
  }
}

// ---------------------------------------------------------------------------
extern "C" void kernel_launch(void* const* d_in, const int* in_sizes, int n_in,
                              void* d_out, int out_size, void* d_ws, size_t ws_size,
                              hipStream_t stream) {
  (void)in_sizes; (void)n_in; (void)out_size; (void)ws_size;
  const float* inputs = (const float*)d_in[0];   // (64,1024,256) f32
  const float* emb    = (const float*)d_in[1];   // (1024,256) f32

  char* ws = (char*)d_ws;
  __bf16* ehi     = (__bf16*)(ws);                       // 512 KiB
  __bf16* elo     = (__bf16*)(ws + 524288);              // 512 KiB
  float*  enorm   = (float*) (ws + 1048576);             // 4 KiB
  float*  partials= (float*) (ws + 1048576 + 4096);      // 2 KiB

  float* out      = (float*)d_out;
  float* out_q    = out;                 // quantized: 16777216 floats
  float* out_loss = out + Q_ELEMS;       // loss: 1 float
  float* out_idx  = out + Q_ELEMS + 1;   // indices-as-float: 65536

  vq_prep  <<<NUM_CODES, DIM, 0, stream>>>(emb, ehi, elo, enorm);
  vq_main  <<<MAIN_BLOCKS, 256, 0, stream>>>(inputs, emb, ehi, elo, enorm,
                                             out_q, out_idx, partials);
  vq_finish<<<1, 32, 0, stream>>>(partials, out_loss);
}
